// GAFE_5849745457918
// MI455X (gfx1250) — matrix operand order repaired and checked by reading, not compiled, not run on hardware
//
#include <hip/hip_runtime.h>
#include <hip/hip_bf16.h>

// ---------------------------------------------------------------------------
// CDNA5 (gfx1250) WMMA implementation of the gated attention + conv block.
// All GEMM math on v_wmma_f32_16x16x32_bf16 (fp32 accumulate), with
// fragment-contiguous LDS/global layouts so operand loads are b128 vectors.
// Shapes: B=8, C=256, H=W=48, P=2304.
// ---------------------------------------------------------------------------

typedef __attribute__((ext_vector_type(16))) __bf16 v16bf;
typedef __attribute__((ext_vector_type(8)))  __bf16 v8bf;
typedef __attribute__((ext_vector_type(8)))  float  v8f;

#define C_CH 256
#define P_PIX 2304
#define HWDIM 48
#define BN_EPS 1e-5f

// native f32 -> bf16 conversion (v_cvt_*bf16* on gfx1250)
__device__ __forceinline__ __bf16 f2bf(float f) { return (__bf16)f; }

__device__ __forceinline__ v8f wmma_bf16(v16bf a, v16bf b, v8f c) {
    return __builtin_amdgcn_wmma_f32_16x16x32_bf16(false, a, false, b,
                                                   (short)0, c, false, false);
}

__device__ __forceinline__ v16bf cat16(v8bf lo, v8bf hi) {
    return __builtin_shufflevector(lo, hi, 0, 1, 2, 3, 4, 5, 6, 7,
                                   8, 9, 10, 11, 12, 13, 14, 15);
}

// A-fragment (16x32 bf16) from row-major [m][k] source, leading dim `ld`
// (elements, multiple of 8; base 16B-aligned). Per ISA 7.12.2 the per-lane
// data is two contiguous 8-element runs: k = kb..kb+7 and kb+16..kb+23.
__device__ __forceinline__ v16bf load_a_rm(const __bf16* p, int ld) {
    int lane = threadIdx.x & 31;
    int m = lane & 15, kb = (lane >> 4) * 8;
    const __bf16* q = p + (size_t)m * ld + kb;
    v8bf lo = *(const v8bf*)(q);
    v8bf hi = *(const v8bf*)(q + 16);
    return cat16(lo, hi);
}

// B-fragment (32x16 bf16) from a K-swizzled [n][k] source, leading dim `ld`
// (elements, multiple of 8). Per-lane data is two contiguous 8-element runs:
// k = kb..kb+7 and kb+16..kb+23 -> two b128 loads.
__device__ __forceinline__ v16bf load_b_sw(const __bf16* p, int ld) {
    int lane = threadIdx.x & 31;
    int n = lane & 15, kb = (lane >> 4) * 8;
    const __bf16* q = p + (size_t)n * ld + kb;
    v8bf lo = *(const v8bf*)(q);
    v8bf hi = *(const v8bf*)(q + 16);
    return cat16(lo, hi);
}

// ---------------------------------------------------------------------------
// Kernel 0: one-time weight repack to bf16 (and tap-major for the convs).
//   Wb  [3][256][256]        : q/k/v 1x1 weights
//   Wp1 [9][256][256]        : conv1 weights, (tap*256+m)*256+c
//   Wp2 [9][256][512]        : conv2 weights, (tap*256+m)*512+c
// ---------------------------------------------------------------------------
__global__ __launch_bounds__(256)
void repack_kernel(const float* __restrict__ wq, const float* __restrict__ wk,
                   const float* __restrict__ wv, const float* __restrict__ w1,
                   const float* __restrict__ w2, __bf16* __restrict__ Wb,
                   __bf16* __restrict__ Wp1, __bf16* __restrict__ Wp2) {
    int idx = blockIdx.x * 256 + threadIdx.x;
    if (idx < 3 * 65536) {
        const float* src = (idx < 65536) ? wq : (idx < 131072 ? wk : wv);
        Wb[idx] = f2bf(src[idx & 65535]);
        return;
    }
    idx -= 3 * 65536;
    if (idx < 9 * 65536) {   // conv1: 256x256x3x3
        int c = idx & 255, m = (idx >> 8) & 255, tap = idx >> 16;
        Wp1[idx] = f2bf(w1[(size_t)(m * 256 + c) * 9 + tap]);
        return;
    }
    idx -= 9 * 65536;
    if (idx < 9 * 131072) {  // conv2: 256x512x3x3
        int c = idx & 511, m = (idx >> 9) & 255, tap = idx >> 17;
        Wp2[idx] = f2bf(w2[(size_t)(m * 512 + c) * 9 + tap]);
    }
}

// ---------------------------------------------------------------------------
// Kernel 1: Q/K/V projection.  out = W[256x256] @ x[256,P] + bias.
// grid (P/64, 3, B), block 256.  A-frags direct from global bf16 weights;
// activations staged K-swizzled in LDS.  q written transposed [P][C].
// ---------------------------------------------------------------------------
__global__ __launch_bounds__(256)
void qkv_kernel(const float* __restrict__ x, const __bf16* __restrict__ Wb,
                const float* __restrict__ bq, const float* __restrict__ bk,
                const float* __restrict__ bv, __bf16* __restrict__ qT,
                __bf16* __restrict__ kO, __bf16* __restrict__ vO) {
    __shared__ __bf16 Xsw[64 * 40];   // [n][k] swizzled, 5120 B

    const int mode = blockIdx.y;
    const __bf16* W   = Wb + (size_t)mode * 65536;
    const float* bias = (mode == 0) ? bq : (mode == 1 ? bk : bv);

    const int b = blockIdx.z, j0 = blockIdx.x * 64;
    const int tid = threadIdx.x, w = tid >> 5, lane = tid & 31;

    v8f acc[2][4] = {};

    for (int k0 = 0; k0 < C_CH; k0 += 32) {
        for (int i = tid; i < 32 * 64; i += 256) {
            int kk = i >> 6, n = i & 63;
            Xsw[n * 40 + kk] =
                f2bf(x[((size_t)b * C_CH + k0 + kk) * P_PIX + j0 + n]);
        }
        __syncthreads();
        v16bf a0 = load_a_rm(W + (size_t)(w * 32 + 0)  * C_CH + k0, C_CH);
        v16bf a1 = load_a_rm(W + (size_t)(w * 32 + 16) * C_CH + k0, C_CH);
#pragma unroll
        for (int jt = 0; jt < 4; ++jt) {
            v16bf bf = load_b_sw(&Xsw[(jt * 16) * 40], 40);
            acc[0][jt] = wmma_bf16(a0, bf, acc[0][jt]);
            acc[1][jt] = wmma_bf16(a1, bf, acc[1][jt]);
        }
        __syncthreads();
    }

    const int nn = lane & 15, half = lane >> 4;
    if (mode == 0) {
#pragma unroll
        for (int s = 0; s < 2; ++s)
#pragma unroll
            for (int jt = 0; jt < 4; ++jt) {
                int m0  = w * 32 + s * 16 + half * 8;
                int col = j0 + jt * 16 + nn;
                v8bf pack;
#pragma unroll
                for (int r = 0; r < 8; ++r)
                    pack[r] = f2bf(acc[s][jt][r] + bias[m0 + r]);
                *(v8bf*)&qT[((size_t)b * P_PIX + col) * C_CH + m0] = pack;
            }
    } else {
        __bf16* o = (mode == 1) ? kO : vO;
#pragma unroll
        for (int s = 0; s < 2; ++s)
#pragma unroll
            for (int jt = 0; jt < 4; ++jt)
#pragma unroll
                for (int r = 0; r < 8; ++r) {
                    int m   = w * 32 + s * 16 + r + half * 8;
                    int col = j0 + jt * 16 + nn;
                    o[((size_t)b * C_CH + m) * P_PIX + col] =
                        f2bf(acc[s][jt][r] + bias[m]);
                }
    }
}

// ---------------------------------------------------------------------------
// Kernel 2: flash attention + residual.  grid (P/64, 1, B), block 256.
// Scores never leave LDS; K-tile resident; all LDS tiles K-swizzled.
// ---------------------------------------------------------------------------
__global__ __launch_bounds__(256)
void attn_kernel(const __bf16* __restrict__ qT, const __bf16* __restrict__ kM,
                 const __bf16* __restrict__ vM, const float* __restrict__ ftr,
                 const float* __restrict__ delta, float* __restrict__ out1) {
    __shared__ __bf16 Ksw[64 * 264];  // [j][c]  33792 B
    __shared__ float  Ssw[64 * 72];   // [j][i]  18432 B (v8f-aligned rows)
    __shared__ __bf16 Psw[64 * 72];   // [j][i]   9216 B
    __shared__ float  mstat[64], lstat[64], rsc[64];
    __shared__ float  red[4 * 64];

    const int b = blockIdx.z, j0 = blockIdx.x * 64;
    const int tid = threadIdx.x, w = tid >> 5, lane = tid & 31;
    const int col = tid & 63, quad = tid >> 6;
    const int nn = lane & 15, hh = lane >> 4;

    for (int i = tid; i < 256 * 64; i += 256) {
        int c = i >> 6, n = i & 63;
        Ksw[n * 264 + c] = kM[((size_t)b * C_CH + c) * P_PIX + j0 + n];
    }
    if (tid < 64) { mstat[tid] = -1e30f; lstat[tid] = 0.f; }

    v8f acc[2][4] = {};
    __syncthreads();

    for (int i0 = 0; i0 < P_PIX; i0 += 64) {
        // ---- S = q^T k, 64x64 tile (2 subtiles per wave) ----
#pragma unroll
        for (int t = 0; t < 2; ++t) {
            int s = w * 2 + t, is = s >> 2, js = s & 3;
            v8f sc = {};
#pragma unroll
            for (int c0 = 0; c0 < C_CH; c0 += 32) {
                v16bf af = load_a_rm(
                    qT + ((size_t)b * P_PIX + i0 + is * 16) * C_CH + c0, C_CH);
                v16bf bf = load_b_sw(&Ksw[(js * 16) * 264 + c0], 264);
                sc = wmma_bf16(af, bf, sc);
            }
            // one contiguous 32B store per accumulator
            *(v8f*)&Ssw[(js * 16 + nn) * 72 + is * 16 + hh * 8] = sc;
        }
        __syncthreads();

        // ---- online softmax over i, per column j ----
        const float* scol = &Ssw[col * 72];
        float mx = -1e30f;
#pragma unroll
        for (int r = 0; r < 16; ++r) mx = fmaxf(mx, scol[quad * 16 + r]);
        red[quad * 64 + col] = mx;
        __syncthreads();
        if (tid < 64) {
            float bm = fmaxf(fmaxf(red[col], red[64 + col]),
                             fmaxf(red[128 + col], red[192 + col]));
            float mo = mstat[col];
            float mn = fmaxf(mo, bm);
            mstat[col] = mn;
            rsc[col]   = __expf(mo - mn);
        }
        __syncthreads();
        float mnew = mstat[col];
        float psum = 0.f;
#pragma unroll
        for (int r = 0; r < 16; ++r) {
            float p = __expf(scol[quad * 16 + r] - mnew);
            Psw[col * 72 + quad * 16 + r] = f2bf(p);   // contiguous per thread
            psum += p;
        }
        red[quad * 64 + col] = psum;
        __syncthreads();
        if (tid < 64)
            lstat[col] = lstat[col] * rsc[col] +
                         red[col] + red[64 + col] + red[128 + col] + red[192 + col];
        __syncthreads();

        // ---- G = G*rescale + V @ P ----
#pragma unroll
        for (int cs = 0; cs < 2; ++cs)
#pragma unroll
            for (int js = 0; js < 4; ++js) {
                float r = rsc[js * 16 + nn];
#pragma unroll
                for (int e = 0; e < 8; ++e) acc[cs][js][e] *= r;
            }
#pragma unroll
        for (int kc = 0; kc < 2; ++kc) {
            v16bf a0 = load_a_rm(
                vM + ((size_t)b * C_CH + w * 32 + 0)  * P_PIX + i0 + kc * 32, P_PIX);
            v16bf a1 = load_a_rm(
                vM + ((size_t)b * C_CH + w * 32 + 16) * P_PIX + i0 + kc * 32, P_PIX);
#pragma unroll
            for (int js = 0; js < 4; ++js) {
                v16bf bf = load_b_sw(&Psw[(js * 16) * 72 + kc * 32], 72);
                acc[0][js] = wmma_bf16(a0, bf, acc[0][js]);
                acc[1][js] = wmma_bf16(a1, bf, acc[1][js]);
            }
        }
        __syncthreads();
    }

    // ---- finalize: out1 = delta * G/l + ftr ----
    const float dl = delta[0];
#pragma unroll
    for (int cs = 0; cs < 2; ++cs)
#pragma unroll
        for (int js = 0; js < 4; ++js) {
            float linv = 1.f / lstat[js * 16 + nn];
#pragma unroll
            for (int r = 0; r < 8; ++r) {
                int c = w * 32 + cs * 16 + r + hh * 8;
                int p = j0 + js * 16 + nn;
                size_t idx = ((size_t)b * C_CH + c) * P_PIX + p;
                out1[idx] = dl * acc[cs][js][r] * linv + ftr[idx];
            }
        }
}

// ---------------------------------------------------------------------------
// Kernel 3: 3x3 avg (count_include_pad) / max pooling of ftr.
// ---------------------------------------------------------------------------
__global__ __launch_bounds__(256)
void pool_kernel(const float* __restrict__ ftr, float* __restrict__ avgp,
                 float* __restrict__ maxp, int total) {
    int idx = blockIdx.x * 256 + threadIdx.x;
    if (idx >= total) return;
    int wp = idx % HWDIM;
    int hp = (idx / HWDIM) % HWDIM;
    int bc = idx / (HWDIM * HWDIM);
    const float* base = ftr + (size_t)bc * (HWDIM * HWDIM);
    float s = 0.f, mx = -1e30f;
#pragma unroll
    for (int dh = -1; dh <= 1; ++dh) {
        int hhp = hp + dh;
        if (hhp < 0 || hhp >= HWDIM) continue;
#pragma unroll
        for (int dw = -1; dw <= 1; ++dw) {
            int ww = wp + dw;
            if (ww < 0 || ww >= HWDIM) continue;
            float v = base[hhp * HWDIM + ww];
            s += v;
            mx = fmaxf(mx, v);
        }
    }
    avgp[idx] = s * (1.f / 9.f);
    maxp[idx] = mx;
}

// ---------------------------------------------------------------------------
// Kernel 4: conv3x3(256->256) + BN + ReLU + sigmoid gate -> gated concat.
// Implicit GEMM as 9 shifted 1x1 GEMMs; grid (48 rows, 1, B), block 256.
// ---------------------------------------------------------------------------
__global__ __launch_bounds__(256)
void conv1_kernel(const float* __restrict__ in, const __bf16* __restrict__ Wp,
                  const float* __restrict__ g1, const float* __restrict__ b1,
                  const float* __restrict__ m1, const float* __restrict__ v1,
                  const float* __restrict__ avgp, const float* __restrict__ maxp,
                  __bf16* __restrict__ cat) {
    __shared__ __bf16 Xsw[50 * 40];   // [padded col][k] swizzled, 4000 B

    const int b = blockIdx.z, h = blockIdx.x;
    const int tid = threadIdx.x, w = tid >> 5, lane = tid & 31;
    v8f acc[2][3] = {};

    for (int kh = 0; kh < 3; ++kh) {
        int hh = h + kh - 1;
        if (hh < 0 || hh >= HWDIM) continue;          // uniform across block
        for (int c0 = 0; c0 < C_CH; c0 += 32) {
            for (int i = tid; i < 32 * 50; i += 256) {
                int kk = i / 50, cc = i % 50;
                float v = 0.f;
                if (cc >= 1 && cc <= HWDIM)
                    v = in[(((size_t)b * C_CH + c0 + kk) * HWDIM + hh) * HWDIM +
                           (cc - 1)];
                Xsw[cc * 40 + kk] = f2bf(v);
            }
            __syncthreads();
#pragma unroll
            for (int kw = 0; kw < 3; ++kw) {
                int tap = kh * 3 + kw;
                v16bf a0 = load_a_rm(
                    Wp + ((size_t)tap * 256 + w * 32 + 0)  * C_CH + c0, C_CH);
                v16bf a1 = load_a_rm(
                    Wp + ((size_t)tap * 256 + w * 32 + 16) * C_CH + c0, C_CH);
#pragma unroll
                for (int js = 0; js < 3; ++js) {
                    v16bf bf = load_b_sw(&Xsw[(js * 16 + kw) * 40], 40);
                    acc[0][js] = wmma_bf16(a0, bf, acc[0][js]);
                    acc[1][js] = wmma_bf16(a1, bf, acc[1][js]);
                }
            }
            __syncthreads();
        }
    }

    const int nn = lane & 15, half = lane >> 4;
#pragma unroll
    for (int s = 0; s < 2; ++s)
#pragma unroll
        for (int js = 0; js < 3; ++js)
#pragma unroll
            for (int r = 0; r < 8; ++r) {
                int m    = w * 32 + s * 16 + r + half * 8;
                int wcol = js * 16 + nn;
                float inv = g1[m] * rsqrtf(v1[m] + BN_EPS);
                float sh  = b1[m] - m1[m] * inv;
                float o   = fmaxf(acc[s][js][r] * inv + sh, 0.f);
                float gate = 1.f / (1.f + __expf(-o));
                size_t p  = ((size_t)b * C_CH + m) * P_PIX + h * HWDIM + wcol;
                size_t pc = ((size_t)b * 512 + m) * P_PIX + h * HWDIM + wcol;
                cat[pc]                        = f2bf(maxp[p] * gate);
                cat[pc + (size_t)C_CH * P_PIX] = f2bf(avgp[p] * gate);
            }
}

// ---------------------------------------------------------------------------
// Kernel 5: conv3x3(512->256) on bf16 cat + BN + ReLU -> final fp32 output.
// ---------------------------------------------------------------------------
__global__ __launch_bounds__(256)
void conv2_kernel(const __bf16* __restrict__ in, const __bf16* __restrict__ Wp,
                  const float* __restrict__ g2, const float* __restrict__ b2,
                  const float* __restrict__ m2, const float* __restrict__ v2,
                  float* __restrict__ out) {
    __shared__ __bf16 Xsw[50 * 40];

    const int b = blockIdx.z, h = blockIdx.x;
    const int tid = threadIdx.x, w = tid >> 5, lane = tid & 31;
    v8f acc[2][3] = {};

    for (int kh = 0; kh < 3; ++kh) {
        int hh = h + kh - 1;
        if (hh < 0 || hh >= HWDIM) continue;
        for (int c0 = 0; c0 < 512; c0 += 32) {
            for (int i = tid; i < 32 * 50; i += 256) {
                int kk = i / 50, cc = i % 50;
                __bf16 v = __builtin_bit_cast(__bf16, (unsigned short)0);
                if (cc >= 1 && cc <= HWDIM)
                    v = in[(((size_t)b * 512 + c0 + kk) * HWDIM + hh) * HWDIM +
                           (cc - 1)];
                Xsw[cc * 40 + kk] = v;
            }
            __syncthreads();
#pragma unroll
            for (int kw = 0; kw < 3; ++kw) {
                int tap = kh * 3 + kw;
                v16bf a0 = load_a_rm(
                    Wp + ((size_t)tap * 256 + w * 32 + 0)  * 512 + c0, 512);
                v16bf a1 = load_a_rm(
                    Wp + ((size_t)tap * 256 + w * 32 + 16) * 512 + c0, 512);
#pragma unroll
                for (int js = 0; js < 3; ++js) {
                    v16bf bf = load_b_sw(&Xsw[(js * 16 + kw) * 40], 40);
                    acc[0][js] = wmma_bf16(a0, bf, acc[0][js]);
                    acc[1][js] = wmma_bf16(a1, bf, acc[1][js]);
                }
            }
            __syncthreads();
        }
    }

    const int nn = lane & 15, half = lane >> 4;
#pragma unroll
    for (int s = 0; s < 2; ++s)
#pragma unroll
        for (int js = 0; js < 3; ++js)
#pragma unroll
            for (int r = 0; r < 8; ++r) {
                int m    = w * 32 + s * 16 + r + half * 8;
                int wcol = js * 16 + nn;
                float inv = g2[m] * rsqrtf(v2[m] + BN_EPS);
                float sh  = b2[m] - m2[m] * inv;
                float o   = fmaxf(acc[s][js][r] * inv + sh, 0.f);
                out[((size_t)b * C_CH + m) * P_PIX + h * HWDIM + wcol] = o;
            }
}

// ---------------------------------------------------------------------------
extern "C" void kernel_launch(void* const* d_in, const int* in_sizes, int n_in,
                              void* d_out, int out_size, void* d_ws, size_t ws_size,
                              hipStream_t stream) {
    const float* ftr    = (const float*)d_in[0];
    const float* wq     = (const float*)d_in[1];
    const float* bq     = (const float*)d_in[2];
    const float* wk     = (const float*)d_in[3];
    const float* bk     = (const float*)d_in[4];
    const float* wv     = (const float*)d_in[5];
    const float* bv     = (const float*)d_in[6];
    const float* delta  = (const float*)d_in[7];
    const float* w_rbc  = (const float*)d_in[8];
    const float* g1     = (const float*)d_in[9];
    const float* b1     = (const float*)d_in[10];
    const float* m1     = (const float*)d_in[11];
    const float* v1     = (const float*)d_in[12];
    const float* w_rbc2 = (const float*)d_in[13];
    const float* g2     = (const float*)d_in[14];
    const float* b2     = (const float*)d_in[15];
    const float* m2     = (const float*)d_in[16];
    const float* v2     = (const float*)d_in[17];

    // workspace layout (bytes): ~107 MB total, fits easily in the 192 MB L2
    char* ws = (char*)d_ws;
    const size_t SZ_BCP_BF = (size_t)8 * C_CH * P_PIX * sizeof(__bf16);  // 9.44 MB
    const size_t SZ_BCP_F  = (size_t)8 * C_CH * P_PIX * sizeof(float);   // 18.9 MB
    size_t off = 0;
    __bf16* qT   = (__bf16*)(ws + off); off += SZ_BCP_BF;
    __bf16* kM   = (__bf16*)(ws + off); off += SZ_BCP_BF;
    __bf16* vM   = (__bf16*)(ws + off); off += SZ_BCP_BF;
    float*  out1 = (float*) (ws + off); off += SZ_BCP_F;
    float*  avgp = (float*) (ws + off); off += SZ_BCP_F;
    float*  maxp = (float*) (ws + off); off += SZ_BCP_F;
    __bf16* cat  = (__bf16*)(ws + off); off += SZ_BCP_F;                 // 18.9 MB bf16 [B][512][P]
    __bf16* Wb   = (__bf16*)(ws + off); off += (size_t)3 * 65536 * 2;
    __bf16* Wp1  = (__bf16*)(ws + off); off += (size_t)9 * 65536 * 2;
    __bf16* Wp2  = (__bf16*)(ws + off); off += (size_t)9 * 131072 * 2;

    // 0) weight repack (bf16, conv weights tap-major)
    int rp_total = 3 * 65536 + 9 * 65536 + 9 * 131072;
    repack_kernel<<<(rp_total + 255) / 256, 256, 0, stream>>>(
        wq, wk, wv, w_rbc, w_rbc2, Wb, Wp1, Wp2);

    // 1) Q/K/V projections (WMMA GEMM)
    qkv_kernel<<<dim3(P_PIX / 64, 3, 8), 256, 0, stream>>>(
        ftr, Wb, bq, bk, bv, qT, kM, vM);

    // 2) flash attention + residual (WMMA)
    attn_kernel<<<dim3(P_PIX / 64, 1, 8), 256, 0, stream>>>(
        qT, kM, vM, ftr, delta, out1);

    // 3) pooling of ftr
    int total = 8 * C_CH * HWDIM * HWDIM;
    pool_kernel<<<(total + 255) / 256, 256, 0, stream>>>(ftr, avgp, maxp, total);

    // 4) conv3x3 + BN + ReLU + sigmoid gate + gated concat (WMMA implicit GEMM)
    conv1_kernel<<<dim3(HWDIM, 1, 8), 256, 0, stream>>>(
        out1, Wp1, g1, b1, m1, v1, avgp, maxp, cat);

    // 5) conv3x3(512->256) + BN + ReLU -> final output (WMMA implicit GEMM)
    conv2_kernel<<<dim3(HWDIM, 1, 8), 256, 0, stream>>>(
        cat, Wp2, g2, b2, m2, v2, (float*)d_out);
}
